// AtenMmQuint8_18940805775746
// MI455X (gfx1250) — compile-verified, matrix operand order checked by reading, not run on hardware
//
#include <hip/hip_runtime.h>
#include <hip/hip_bf16.h>

typedef __attribute__((ext_vector_type(8))) int v8i;
typedef __attribute__((ext_vector_type(4))) int v4i;

#define M_DIM 2048
#define N_DIM 2048
#define K_DIM 4096

// ---------------------------------------------------------------------------
// 8-bit WMMA fragment layout (CDNA5 ISA 7.12.2), 16x64 tile = 1024 bytes:
//   lane = row + ((k>>3)&1)*16
//   vgpr = (k>>5)*4 + ((k>>4)&1)*2 + ((k&7)>>2), byte-in-dword = k&3
// In-tile byte address (two 512B halves so ds_load_b128 sweeps all 64 banks):
//   off = (vgpr>>2)*512 + lane*16 + (vgpr&3)*4 + (k&3)
// Fragment read: b128 at (base + lane*16) and (base + 512 + lane*16).
// Tiles are stored kt-major-contiguous: tile(row_tile, kt) at (row_tile*64+kt)*1024.
// ---------------------------------------------------------------------------

__global__ __launch_bounds__(256) void pack_x_kernel(const int* __restrict__ x,
                                                     unsigned char* __restrict__ xq,
                                                     int* __restrict__ rowsum) {
    const int m = blockIdx.x;            // one row per block
    const int t = threadIdx.x;           // 0..255
    const int mt = m >> 4;
    const int m_local = m & 15;
    const int* xrow = x + (size_t)m * K_DIM;
    unsigned int* xq32 = (unsigned int*)xq;

    int sum = 0;
    #pragma unroll
    for (int g = 0; g < 4; ++g) {
        const int k0 = (t + g * 256) << 2;            // dword-aligned k
        v4i xv = *(const v4i*)(xrow + k0);            // coalesced b128 read
        sum += xv.x + xv.y + xv.z + xv.w;
        unsigned int packed = (unsigned)(xv.x & 255)
                            | ((unsigned)(xv.y & 255) << 8)
                            | ((unsigned)(xv.z & 255) << 16)
                            | ((unsigned)(xv.w & 255) << 24);
        const int kt = k0 >> 6;
        const int kl = k0 & 63;
        const int lane = m_local + (((kl >> 3) & 1) << 4);
        const int vgpr = ((kl >> 5) << 2) + (((kl >> 4) & 1) << 1) + ((kl & 7) >> 2);
        const int off = ((mt * 64 + kt) << 10)
                      + ((vgpr >> 2) << 9) + (lane << 4) + ((vgpr & 3) << 2);
        xq32[off >> 2] = packed;
    }

    __shared__ int red[256];
    red[t] = sum;
    __syncthreads();
    #pragma unroll
    for (int s = 128; s > 0; s >>= 1) {
        if (t < s) red[t] += red[t + s];
        __syncthreads();
    }
    if (t == 0) rowsum[m] = red[0];
}

__global__ __launch_bounds__(256) void pack_y_kernel(const int* __restrict__ y,
                                                     unsigned char* __restrict__ yq) {
    const int bid = blockIdx.x;          // 0..(128*64-1)
    const int nt = bid >> 6;             // N tile (16 cols)
    const int kt = bid & 63;             // K tile (64 rows)
    const int t = threadIdx.x;
    const int n_local = t & 15;
    const int k0 = (t >> 4) << 2;        // 0,4,...,60 (aligned 4 => one dword)
    const int n = nt * 16 + n_local;
    const int tile_base = (nt * 64 + kt) << 10;

    unsigned int packed = 0;
    #pragma unroll
    for (int i = 0; i < 4; ++i) {
        const int v = y[(size_t)(kt * 64 + k0 + i) * N_DIM + n];  // coalesced in n
        packed |= (unsigned)(v & 255) << (8 * i);
    }
    const int lane = n_local + (((k0 >> 3) & 1) << 4);
    const int vgpr = ((k0 >> 5) << 2) + (((k0 >> 4) & 1) << 1) + ((k0 & 7) >> 2);
    const int off = tile_base + ((vgpr >> 2) << 9) + (lane << 4) + ((vgpr & 3) << 2);
    *(unsigned int*)(yq + off) = packed;
}

__global__ __launch_bounds__(256) void colsum_kernel(const int* __restrict__ y,
                                                     int* __restrict__ colsum) {
    const int n = blockIdx.x * 256 + threadIdx.x;
    int sum = 0;
    for (int k = 0; k < K_DIM; ++k) sum += y[(size_t)k * N_DIM + n];  // coalesced
    colsum[n] = sum;
}

// One async b128 copy: LDS[lds_off ...+15] = MEM[gptr ...+15]  (ASYNCcnt)
#define ASYNC_CP_B128(lds_off, gptr)                                      \
    asm volatile("global_load_async_to_lds_b128 %0, %1, off"              \
                 :: "v"(lds_off), "v"(gptr) : "memory")

__global__ __launch_bounds__(256) void gemm_iu8_kernel(const unsigned char* __restrict__ xq,
                                                       const unsigned char* __restrict__ yq,
                                                       const int* __restrict__ rowsum,
                                                       const int* __restrict__ colsum,
                                                       float* __restrict__ out) {
    // [buf][A: 8 tiles | B: 8 tiles], 16KB per stage, double buffered
    __shared__ __attribute__((aligned(128))) unsigned char lds[2][16384];

    const int t      = threadIdx.x;
    const int lane   = t & 31;
    const int w      = t >> 5;                 // 8 waves
    const int wave_m = w >> 2;                 // 0..1
    const int wave_n = w & 3;                  // 0..3
    const int mtBlk  = blockIdx.y * 8;         // block's first M tile
    const int ntBlk  = blockIdx.x * 8;         // block's first N tile
    const int mt0    = wave_m * 4;             // wave's tiles within block
    const int nt0    = wave_n * 2;

    // --- staging plan: 256 threads x 64B = 16KB per stage; chunk c = 1KB tile ---
    const int c = t >> 4;                      // 0..15 (0-7 = A tiles, 8-15 = B tiles)
    const int o = (t & 15) * 64;               // byte offset within tile
    const unsigned char* gsrc =
        (c < 8) ? (xq + ((size_t)(mtBlk + c) * 64) * 1024 + o)
                : (yq + ((size_t)(ntBlk + (c - 8)) * 64) * 1024 + o);
    const unsigned lds_base = (unsigned)(size_t)&lds[0][c * 1024 + o];

    const v8i vzero = {0, 0, 0, 0, 0, 0, 0, 0};
    v8i acc[4][2];
    #pragma unroll
    for (int i = 0; i < 4; ++i)
        #pragma unroll
        for (int j = 0; j < 2; ++j)
            acc[i][j] = vzero;

    const int KT = K_DIM / 64;   // 64 k-steps

    // Prologue: stage kt=0 into buffer 0 (4 async b128 per thread)
    #pragma unroll
    for (int q = 0; q < 4; ++q)
        ASYNC_CP_B128(lds_base + q * 16, gsrc + q * 16);

    for (int kt = 0; kt < KT; ++kt) {
        const int b = kt & 1;
        __syncthreads();   // previous compute done -> safe to overwrite buf[b^1]
        if (kt + 1 < KT) {
            const unsigned char* gn = gsrc + (size_t)(kt + 1) * 1024;
            const unsigned ln = lds_base + ((kt + 1) & 1) * 16384u;
            #pragma unroll
            for (int q = 0; q < 4; ++q)
                ASYNC_CP_B128(ln + q * 16, gn + q * 16);
            // async loads complete in order: <=4 left means stage kt is done
            asm volatile("s_wait_asynccnt 0x4" ::: "memory");
        } else {
            asm volatile("s_wait_asynccnt 0x0" ::: "memory");
        }
        __syncthreads();   // buf[b] published by all waves

        v8i a[4], bb[2];
        #pragma unroll
        for (int i = 0; i < 4; ++i) {
            const unsigned char* p = &lds[b][(mt0 + i) * 1024];
            union { v8i v; v4i h[2]; } u;
            u.h[0] = *(const v4i*)(p + lane * 16);          // ds_load_b128
            u.h[1] = *(const v4i*)(p + 512 + lane * 16);    // ds_load_b128
            a[i] = u.v;
        }
        #pragma unroll
        for (int j = 0; j < 2; ++j) {
            const unsigned char* p = &lds[b][8192 + (nt0 + j) * 1024];
            union { v8i v; v4i h[2]; } u;
            u.h[0] = *(const v4i*)(p + lane * 16);
            u.h[1] = *(const v4i*)(p + 512 + lane * 16);
            bb[j] = u.v;
        }
        #pragma unroll
        for (int i = 0; i < 4; ++i)
            #pragma unroll
            for (int j = 0; j < 2; ++j)
                acc[i][j] = __builtin_amdgcn_wmma_i32_16x16x64_iu8(
                    /*sgn_a=*/false, a[i], /*sgn_b=*/false, bb[j],
                    acc[i][j], /*reuse_a=*/false, /*reuse_b=*/false);
    }

    // Epilogue: out = Sx*Sy * (dot - Y_ZP*rowsum - X_ZP*colsum + K*X_ZP*Y_ZP)
    const float scale = 0.199f * 0.0215f;
    const int CONST_TERM = K_DIM * 65 * 160;     // 42,598,400 (fits int32)
    #pragma unroll
    for (int i = 0; i < 4; ++i) {
        const int mbase = (mtBlk + mt0 + i) * 16 + (lane >> 4) * 8;
        int rs[8];
        #pragma unroll
        for (int r = 0; r < 8; ++r) rs[r] = rowsum[mbase + r];
        #pragma unroll
        for (int j = 0; j < 2; ++j) {
            const int n = (ntBlk + nt0 + j) * 16 + (lane & 15);
            const int cs = colsum[n];
            #pragma unroll
            for (int r = 0; r < 8; ++r) {
                const int v = acc[i][j][r] - 160 * rs[r] - 65 * cs + CONST_TERM;
                out[(size_t)(mbase + r) * N_DIM + n] = scale * (float)v;
            }
        }
    }
}

extern "C" void kernel_launch(void* const* d_in, const int* in_sizes, int n_in,
                              void* d_out, int out_size, void* d_ws, size_t ws_size,
                              hipStream_t stream) {
    const int* x = (const int*)d_in[0];   // [2048, 4096] values in [0,256)
    const int* y = (const int*)d_in[1];   // [4096, 2048] values in [0,256)
    float* out = (float*)d_out;           // [2048, 2048] fp32

    unsigned char* xq = (unsigned char*)d_ws;                 // 8 MB
    unsigned char* yq = xq + (size_t)M_DIM * K_DIM;           // 8 MB
    int* rowsum = (int*)(yq + (size_t)K_DIM * N_DIM);         // 8 KB
    int* colsum = rowsum + M_DIM;                             // 8 KB

    pack_x_kernel<<<M_DIM, 256, 0, stream>>>(x, xq, rowsum);
    pack_y_kernel<<<(N_DIM / 16) * (K_DIM / 64), 256, 0, stream>>>(y, yq);
    colsum_kernel<<<N_DIM / 256, 256, 0, stream>>>(y, colsum);
    gemm_iu8_kernel<<<dim3(N_DIM / 128, M_DIM / 128), 256, 0, stream>>>(
        xq, yq, rowsum, colsum, out);
}
